// GMSNet_50861002719257
// MI455X (gfx1250) — compile-verified, matrix operand order checked by reading, not run on hardware
//
#include <hip/hip_runtime.h>

typedef __attribute__((ext_vector_type(16))) _Float16 v16h;
typedef __attribute__((ext_vector_type(8)))  float    v8f;
typedef long long i64;

// ---------------- k0: zero a float range (grid-stride) ----------------
__global__ void k0_zero(float* __restrict__ p, int n) {
  int i = blockIdx.x * blockDim.x + threadIdx.x;
  int stride = gridDim.x * blockDim.x;
  for (; i < n; i += stride) p[i] = 0.0f;
}

// ---------------- k1: lin + global sum/sumsq for GraphNorm ----------------
__global__ void k1_stats(const float* __restrict__ x,
                         const float* __restrict__ lw,
                         const float* __restrict__ lb,
                         float* __restrict__ stats, int N) {
  float lw00 = lw[0], lw01 = lw[1], lw10 = lw[2], lw11 = lw[3];
  float b0 = lb[0], b1 = lb[1];
  float s0 = 0.f, s1 = 0.f, q0 = 0.f, q1 = 0.f;
  int i = blockIdx.x * blockDim.x + threadIdx.x;
  int stride = gridDim.x * blockDim.x;
  for (; i < N; i += stride) {
    float x0 = x[2 * i], x1 = x[2 * i + 1];
    float h0 = fmaf(x0, lw00, fmaf(x1, lw01, b0));
    float h1 = fmaf(x0, lw10, fmaf(x1, lw11, b1));
    s0 += h0; s1 += h1;
    q0 = fmaf(h0, h0, q0); q1 = fmaf(h1, h1, q1);
  }
#pragma unroll
  for (int off = 16; off > 0; off >>= 1) {
    s0 += __shfl_down(s0, off, 32);
    s1 += __shfl_down(s1, off, 32);
    q0 += __shfl_down(q0, off, 32);
    q1 += __shfl_down(q1, off, 32);
  }
  if ((threadIdx.x & 31) == 0) {
    atomicAdd(&stats[0], s0); atomicAdd(&stats[1], s1);
    atomicAdd(&stats[2], q0); atomicAdd(&stats[3], q1);
  }
}

// ---------------- k2: finalize GraphNorm stats (eps = D = 2) ----------------
__global__ void k2_finalize(float* __restrict__ stats,
                            const float* __restrict__ gn_ms, int N) {
  if (threadIdx.x == 0 && blockIdx.x == 0) {
    float rn = 1.0f / (float)N;
#pragma unroll
    for (int d = 0; d < 2; ++d) {
      float m = stats[d] * rn;
      float msub = m * gn_ms[d];
      float var = stats[2 + d] * rn - 2.0f * m * msub + msub * msub;
      stats[4 + d] = msub;                 // mean * mean_scale
      stats[6 + d] = rsqrtf(var + 2.0f);   // eps quirk: eps = D = 2
    }
  }
}

// ---------------- k3: degree histogram over col (NT index stream) ----------------
__global__ void k3_deg(const i64* __restrict__ ei, float* __restrict__ deg, int NE) {
  const i64* col = ei + NE;
  int i = blockIdx.x * blockDim.x + threadIdx.x;
  int stride = gridDim.x * blockDim.x;
  for (; i < NE; i += stride) {
    __builtin_prefetch(&col[i + stride], 0, 3);
    int c = (int)__builtin_nontemporal_load(&col[i]);   // read-once stream: keep L2 clean
    atomicAdd(&deg[c], 1.0f);
  }
}

// ---------------- k4: node prep: norm+relu, dis, hw, acc = selfloop+bias+residual, counts ----
__global__ void k4_node(const float* __restrict__ x,
                        const float* __restrict__ lw, const float* __restrict__ lb,
                        const float* __restrict__ gnw, const float* __restrict__ gnb,
                        const float* __restrict__ gcw, const float* __restrict__ gcb,
                        const float* __restrict__ stats,
                        const float* __restrict__ degarr,
                        const i64* __restrict__ batch,
                        float* __restrict__ dis, float* __restrict__ hw,
                        float* __restrict__ acc, float* __restrict__ cnt, int N) {
  float lw00 = lw[0], lw01 = lw[1], lw10 = lw[2], lw11 = lw[3];
  float lb0 = lb[0], lb1 = lb[1];
  float ms0 = stats[4], ms1 = stats[5], iv0 = stats[6], iv1 = stats[7];
  float w0 = gnw[0], w1 = gnw[1], nb0 = gnb[0], nb1 = gnb[1];
  float g00 = gcw[0], g01 = gcw[1], g10 = gcw[2], g11 = gcw[3];
  float gb0 = gcb[0], gb1 = gcb[1];
  int i = blockIdx.x * blockDim.x + threadIdx.x;
  int stride = gridDim.x * blockDim.x;
  for (; i < N; i += stride) {
    float x0 = x[2 * i], x1 = x[2 * i + 1];
    float h0 = fmaf(x0, lw00, fmaf(x1, lw01, lb0));
    float h1 = fmaf(x0, lw10, fmaf(x1, lw11, lb1));
    h0 = fmaxf(fmaf(w0 * (h0 - ms0), iv0, nb0), 0.0f);
    h1 = fmaxf(fmaf(w1 * (h1 - ms1), iv1, nb1), 0.0f);
    float dg = degarr[i] + 1.0f;         // +1 self loop
    float di = rsqrtf(dg);
    dis[i] = di;
    float hw0 = fmaf(h0, g00, h1 * g01);
    float hw1 = fmaf(h0, g10, h1 * g11);
    hw[2 * i] = hw0; hw[2 * i + 1] = hw1;
    float idg = di * di;                 // 1/deg
    acc[2 * i]     = fmaf(hw0, idg, gb0) + h0;  // self-loop + bias + residual
    acc[2 * i + 1] = fmaf(hw1, idg, gb1) + h1;
    atomicAdd(&cnt[(int)batch[i]], 1.0f);
  }
}

// ---------------- k5: edge message scatter (NT index streams) ----------------
__global__ void k5_edge(const i64* __restrict__ ei,
                        const float* __restrict__ dis,
                        const float* __restrict__ hw,
                        float* __restrict__ acc, int NE) {
  const i64* rowp = ei;
  const i64* colp = ei + NE;
  const float2* hw2 = (const float2*)hw;
  int i = blockIdx.x * blockDim.x + threadIdx.x;
  int stride = gridDim.x * blockDim.x;
  for (; i < NE; i += stride) {
    __builtin_prefetch(&rowp[i + stride], 0, 3);
    __builtin_prefetch(&colp[i + stride], 0, 3);
    int r = (int)__builtin_nontemporal_load(&rowp[i]);
    int c = (int)__builtin_nontemporal_load(&colp[i]);
    float w = dis[r] * dis[c];           // random gathers: L2-resident (4 MB)
    float2 h = hw2[r];                   // random gather: L2-resident (8 MB)
    atomicAdd(&acc[2 * c],     w * h.x);
    atomicAdd(&acc[2 * c + 1], w * h.y);
  }
}

// ---------------- k6: fused fc1 + InstanceNorm + fc2 (WMMA) + sorted scatter-mean ----------
// Block = 256 threads = 8 waves; wave handles 16 nodes/iter, 32 iters -> 4096 contiguous
// nodes per block. Batch is sorted, so each block touches only a tiny graph range:
// accumulate in LDS bins, flush [gmin,gmax] with global atomics.
__global__ void __launch_bounds__(256)
k6_mlp(const float* __restrict__ acc,
       const float* __restrict__ fc1w, const float* __restrict__ fc1b,
       const float* __restrict__ fc2w, const float* __restrict__ fc2b,
       const i64* __restrict__ batch,
       float* __restrict__ gsum, int N, int G) {
  __shared__ float sW1[128];
  __shared__ float sB1[64];
  __shared__ float sW2[128];
  __shared__ float sB2[2];
  __shared__ float sbins[2048];          // G <= 1024 (reference: G == 1024)

  const int tid  = threadIdx.x;
  const int wave = tid >> 5;
  const int lane = tid & 31;
  const int half = lane >> 4;            // which K-half of the A/B layout
  const int n    = lane & 15;            // matrix row (node-in-tile) / output column

  for (int i = tid; i < 128; i += 256) sW1[i] = fc1w[i];
  for (int i = tid; i < 64;  i += 256) sB1[i] = fc1b[i];
  for (int i = tid; i < 128; i += 256) sW2[i] = fc2w[i];
  if (tid < 2) sB2[tid] = fc2b[tid];
  for (int i = tid; i < 2048; i += 256) sbins[i] = 0.0f;
  __syncthreads();

  const int gclamp = (G < 1024 ? G : 1024) - 1;
  const float outBias = (n < 2) ? fc2b[n] : 0.0f;   // hoisted: no per-element LDS load

  // B matrix (fc2_w^T in columns 0..1, zeros elsewhere). 16x16x32 f16 layout:
  // lane column = n; half 0 holds K {0..7,16..23}, half 1 holds K {8..15,24..31}.
  v16h B1, B2;
#pragma unroll
  for (int i = 0; i < 16; ++i) {
    int K = (i < 8 ? i : 8 + i) + half * 8;
    float w1 = (n < 2) ? sW2[n * 64 + K]      : 0.0f;
    float w2 = (n < 2) ? sW2[n * 64 + K + 32] : 0.0f;
    B1[i] = (_Float16)w1;
    B2[i] = (_Float16)w2;
  }

  const float2* acc2 = (const float2*)acc;
  const int blockStart = blockIdx.x * 4096;
  const int ITERS = 32;
  for (int it = 0; it < ITERS; ++it) {
    const int tilebase = blockStart + (wave + 8 * it) * 16;
    const int node = tilebase + n;
    const int nc = node < N ? node : N - 1;
    const bool valid = node < N;

    float2 hv = acc2[nc];                       // single b64 load per node
    float h0 = hv.x, h1 = hv.y;

    // coalesced per-node batch id (extracting lanes fetch via shuffle below)
    int gme = (int)batch[nc];
    gme = gme < 0 ? 0 : (gme > gclamp ? gclamp : gme);

    // fc1: each lane computes exactly the 32 channels its WMMA A slots need.
    float hh[32];
    float s = 0.0f, q = 0.0f;
#pragma unroll
    for (int i = 0; i < 16; ++i) {
      int c = (i < 8 ? i : 8 + i) + half * 8;        // K for A1 slot i
      float a = fmaf(h0, sW1[2 * c],        fmaf(h1, sW1[2 * c + 1],        sB1[c]));
      float b = fmaf(h0, sW1[2 * (c + 32)], fmaf(h1, sW1[2 * (c + 32) + 1], sB1[c + 32]));
      hh[i] = a; hh[16 + i] = b;
      s += a + b;
      q = fmaf(a, a, fmaf(b, b, q));
    }
    // combine with partner lane (same node, other 32 channels)
    s += __shfl_xor(s, 16, 32);
    q += __shfl_xor(q, 16, 32);
    float m   = s * (1.0f / 64.0f);
    float var = q * (1.0f / 64.0f) - m * m;
    float inv = rsqrtf(var + 64.0f);                 // eps quirk: eps = H = 64
    float scale = valid ? inv : 0.0f;                // mask tail rows (keeps EXEC full)

    v16h A1, A2;
#pragma unroll
    for (int i = 0; i < 16; ++i) {
      A1[i] = (_Float16)((hh[i]      - m) * scale);
      A2[i] = (_Float16)((hh[16 + i] - m) * scale);
    }

    v8f c8 = {};
    c8 = __builtin_amdgcn_wmma_f32_16x16x32_f16(false, A1, false, B1, (short)0, c8, false, false);
    c8 = __builtin_amdgcn_wmma_f32_16x16x32_f16(false, A2, false, B2, (short)0, c8, false, false);

    // D layout: VGPR r, lanes 0-15 -> M=r, lanes 16-31 -> M=8+r. Columns 0,1 are outputs.
    // Shuffles run with full EXEC; only lanes n<2 do the LDS atomic.
#pragma unroll
    for (int r = 0; r < 8; ++r) {
      int g = __shfl(gme, half * 8 + r, 32);   // batch id of node tilebase+half*8+r
      int nd = tilebase + half * 8 + r;
      if (n < 2 && nd < N)
        atomicAdd(&sbins[g * 2 + n], c8[r] + outBias);
    }
  }
  __syncthreads();

  // flush only the graph range this contiguous node span touches (batch sorted)
  int lastNode = blockStart + 4096;
  lastNode = (lastNode < N ? lastNode : N) - 1;
  int gmin = (int)batch[blockStart];
  int gmax = (int)batch[lastNode];
  gmin = gmin < 0 ? 0 : (gmin > gclamp ? gclamp : gmin);
  gmax = gmax < 0 ? 0 : (gmax > gclamp ? gclamp : gmax);
  for (int idx = gmin * 2 + tid; idx <= gmax * 2 + 1; idx += 256)
    atomicAdd(&gsum[idx], sbins[idx]);
}

// ---------------- k7: final divide ----------------
__global__ void k7_out(const float* __restrict__ gsum, const float* __restrict__ cnt,
                       float* __restrict__ out, int G) {
  int i = blockIdx.x * blockDim.x + threadIdx.x;
  if (i < 2 * G) out[i] = gsum[i] / fmaxf(cnt[i >> 1], 1.0f);
}

// ---------------- host launcher ----------------
extern "C" void kernel_launch(void* const* d_in, const int* in_sizes, int n_in,
                              void* d_out, int out_size, void* d_ws, size_t ws_size,
                              hipStream_t stream) {
  const float* x     = (const float*)d_in[0];
  const float* lin_w = (const float*)d_in[1];
  const float* lin_b = (const float*)d_in[2];
  const float* gn_w  = (const float*)d_in[3];
  const float* gn_b  = (const float*)d_in[4];
  const float* gn_ms = (const float*)d_in[5];
  const float* gcn_w = (const float*)d_in[6];
  const float* gcn_b = (const float*)d_in[7];
  const float* fc1w  = (const float*)d_in[8];
  const float* fc1b  = (const float*)d_in[9];
  const float* fc2w  = (const float*)d_in[10];
  const float* fc2b  = (const float*)d_in[11];
  const i64*   ei    = (const i64*)d_in[12];   // int64 edge_index [2, NE]
  const i64*   batch = (const i64*)d_in[13];   // int64, sorted

  const int N  = in_sizes[0] / 2;
  const int NE = in_sizes[12] / 2;
  const int G  = out_size / 2;

  // workspace layout (floats)
  float* ws    = (float*)d_ws;
  float* stats = ws;                       // 16 (8 used)
  float* deg   = ws + 16;                  // N
  float* dis   = deg + (size_t)N;          // N
  float* hw    = dis + (size_t)N;          // 2N
  float* acc   = hw + 2 * (size_t)N;       // 2N
  float* cnt   = acc + 2 * (size_t)N;      // G
  float* gsum  = cnt + (size_t)G;          // 2G

  // zero: stats+deg (contiguous), cnt+gsum (contiguous)
  k0_zero<<<512, 256, 0, stream>>>(stats, 16 + N);
  k0_zero<<<16, 256, 0, stream>>>(cnt, 3 * G);

  k1_stats<<<1024, 256, 0, stream>>>(x, lin_w, lin_b, stats, N);
  k2_finalize<<<1, 32, 0, stream>>>(stats, gn_ms, N);
  k3_deg<<<2048, 256, 0, stream>>>(ei, deg, NE);
  k4_node<<<1024, 256, 0, stream>>>(x, lin_w, lin_b, gn_w, gn_b, gcn_w, gcn_b,
                                    stats, deg, batch, dis, hw, acc, cnt, N);
  k5_edge<<<4096, 256, 0, stream>>>(ei, dis, hw, acc, NE);

  int nblk6 = (N + 4095) / 4096;
  k6_mlp<<<nblk6, 256, 0, stream>>>(acc, fc1w, fc1b, fc2w, fc2b, batch, gsum, N, G);

  k7_out<<<(2 * G + 255) / 256, 256, 0, stream>>>(gsum, cnt, (float*)d_out, G);
}
